// GAT_DSSE_BiLevel_Stable_68685116997813
// MI455X (gfx1250) — matrix operand order repaired and checked by reading, not compiled, not run on hardware
//
#include <hip/hip_runtime.h>
#include <hip/hip_bf16.h>
#include <math.h>

// ---------------------------------------------------------------------------
// GAT_DSSE_BiLevel_Stable for MI455X (gfx1250, wave32, WMMA)
//
//   per layer l in {0,1}:
//     zero(agg, maxnjKey, mKey, sSum)
//     gemm_wmma<K,128>: xl = h @ Wl + bl   (bf16 WMMA, f32 accum, unrolled K)
//     gemm_wmma<K,128>: xr = h @ Wr + br
//     edge_pass1..4: logits -> Lipschitz norm -> segment softmax -> scatter
//     node_finish: +bias, LayerNorm, (+0.1*residual), leakyReLU
//   head: LN -> W1(+bn,lrelu) -> W2(+bn,lrelu) -> W3
// ---------------------------------------------------------------------------

#define NNODES 50000
#define NEDGES 800000

typedef __attribute__((ext_vector_type(16))) __bf16 v16bf;
typedef __attribute__((ext_vector_type(8)))  float  v8f;

// ---- ordered-int keys so unsigned atomicMax == float max (handles negatives)
__device__ __forceinline__ unsigned fkey(float f) {
  unsigned b = __float_as_uint(f);
  return (b & 0x80000000u) ? ~b : (b | 0x80000000u);
}
__device__ __forceinline__ float fkeyinv(unsigned k) {
  unsigned b = (k & 0x80000000u) ? (k & 0x7FFFFFFFu) : ~k;
  return __uint_as_float(b);
}
__device__ __forceinline__ float lrelu(float v) { return v > 0.f ? v : 0.01f * v; }

// ---------------------------------------------------------------------------
// WMMA GEMM: out[M,NCOLS] = epilogue(A[M,K] @ W[K,NCOLS] + bias)
// Block = 256 threads = 8 waves; wave w owns the 16x16 tile at
// cols [blockIdx.y*128 + 16w). The 16xK fp32 A-tile is converted ONCE per
// block into LDS as bf16 pairs pre-packed in the ISA 7.12.2 A-fragment
// layout, so each wave's fragment is one 32-byte contiguous LDS read.
// K-loop fully unrolled -> K/32 back-to-back v_wmma per wave.
// ---------------------------------------------------------------------------
template <int K, int NCOLS, bool BN, bool ACT>
__global__ __launch_bounds__(256)
void gemm_wmma(const float* __restrict__ A, const float* __restrict__ W,
               const float* __restrict__ bias,
               const float* __restrict__ bn_g, const float* __restrict__ bn_b,
               float* __restrict__ out, int M)
{
  // [t = k-step][lane][pair p] ; 256 u32 per k-step
  __shared__ unsigned lds_pk[(K / 32) * 256];

  const int tid  = threadIdx.x;
  const int wave = tid >> 5;
  const int lane = tid & 31;
  const int row0 = blockIdx.x * 16;
  const int colbase = blockIdx.y * 128 + wave * 16;

  // Convert + pack A tile into fragment order (done once per block).
  for (int i = tid; i < 8 * K; i += 256) {
    const int p = i & 7, L = (i >> 3) & 31, t = i >> 8;
    const int m = L & 15, h = L >> 4;
    const int k0 = t * 32 + ((p < 4) ? (2 * p + 8 * h) : (2 * (p - 4) + 16 + 8 * h));
    const float f0 = A[(size_t)(row0 + m) * K + k0];
    const float f1 = A[(size_t)(row0 + m) * K + k0 + 1];
    union { __bf16 h2[2]; unsigned u; } cv;
    cv.h2[0] = (__bf16)f0; cv.h2[1] = (__bf16)f1;
    lds_pk[i] = cv.u;
  }
  __syncthreads();

  const int half = lane >> 4;
  const int n    = lane & 15;

  v8f acc = {0.f, 0.f, 0.f, 0.f, 0.f, 0.f, 0.f, 0.f};
#pragma unroll
  for (int t = 0; t < K / 32; ++t) {
    // A fragment: contiguous 32B per lane (2 x ds_load_b128)
    const v16bf a = *reinterpret_cast<const v16bf*>(&lds_pk[t * 256 + lane * 8]);
    // B 32x16 bf16: e<8 -> K=e+8*half, e>=8 -> K=16+8*half+(e-8); col = lane&15
    v16bf b;
#pragma unroll
    for (int e = 0; e < 16; ++e) {
      const int kb = t * 32 + ((e < 8) ? (e + 8 * half) : (16 + 8 * half + (e - 8)));
      b[e] = (__bf16)W[kb * NCOLS + colbase + n];
    }
    acc = __builtin_amdgcn_wmma_f32_16x16x32_bf16(false, a, false, b,
                                                  (short)0, acc, false, false);
  }

  const float bninv = 0.9999950000374997f; // 1/sqrt(1+1e-5)
  const int col = colbase + n;
#pragma unroll
  for (int r = 0; r < 8; ++r) {
    const int row = row0 + r + 8 * half;   // C/D layout: VGPR r -> M=r (+8 upper)
    float v = acc[r] + bias[col];
    if (BN)  v = v * (bn_g[col] * bninv) + bn_b[col];
    if (ACT) v = lrelu(v);
    out[row * NCOLS + col] = v;
  }
  (void)M;
}

// ---------------------------------------------------------------------------
// Edge pass 1: wave per edge, 4 channels per lane (head = lane>>3).
// e_pre = sum_c lrelu(xi+xj+ea)*att ; ni = ||xi_h||+eps ;
// atomicMax of ||xj_h||+eps into maxnjKey[dst,h].
// ---------------------------------------------------------------------------
__global__ __launch_bounds__(256)
void edge_pass1(const float* __restrict__ xl, const float* __restrict__ xr,
                const int* __restrict__ ei, const float* __restrict__ eattr,
                const float* __restrict__ We, const float* __restrict__ att,
                float* __restrict__ epre, float* __restrict__ niB,
                unsigned* __restrict__ maxnjKey, int En)
{
  __shared__ float sWe[8 * 128];
  __shared__ float sAtt[128];
  const int tid = threadIdx.x;
  for (int i = tid; i < 8 * 128; i += 256) sWe[i] = We[i];
  if (tid < 128) sAtt[tid] = att[tid];
  __syncthreads();

  const int wave = tid >> 5, lane = tid & 31;
  const int e = blockIdx.x * 8 + wave;
  if (e >= En) return;
  // speculative prefetch of the streaming edge_attr a few blocks ahead
  __builtin_prefetch(eattr + (size_t)e * 8 + 4096, 0, 1);

  const int src = ei[e], dst = ei[En + e];
  const int c0 = lane * 4;

  float4 xi4 = *(const float4*)(xr + (size_t)dst * 128 + c0);
  float4 xj4 = *(const float4*)(xl + (size_t)src * 128 + c0);
  float xiv[4] = {xi4.x, xi4.y, xi4.z, xi4.w};
  float xjv[4] = {xj4.x, xj4.y, xj4.z, xj4.w};

  float ed[8];
#pragma unroll
  for (int d = 0; d < 8; ++d) ed[d] = eattr[(size_t)e * 8 + d];

  float pe = 0.f, si = 0.f, sj = 0.f;
#pragma unroll
  for (int j = 0; j < 4; ++j) {
    float ea = 0.f;
#pragma unroll
    for (int d = 0; d < 8; ++d) ea += ed[d] * sWe[d * 128 + c0 + j];
    float z = lrelu(xiv[j] + xjv[j] + ea);
    pe += z * sAtt[c0 + j];
    si += xiv[j] * xiv[j];
    sj += xjv[j] * xjv[j];
  }
  // reduce across the 8 lanes of each head group
  for (int msk = 1; msk < 8; msk <<= 1) {
    pe += __shfl_xor(pe, msk, 32);
    si += __shfl_xor(si, msk, 32);
    sj += __shfl_xor(sj, msk, 32);
  }
  if ((lane & 7) == 0) {
    const int h = lane >> 3;
    epre[(size_t)e * 4 + h] = pe;
    niB [(size_t)e * 4 + h] = sqrtf(si) + 1e-8f;
    const float nj = sqrtf(sj) + 1e-8f;
    atomicMax(&maxnjKey[(size_t)dst * 4 + h], fkey(nj));
  }
}

// Edge pass 2: Lipschitz norm + double clip, atomicMax logit per (dst,head)
__global__ __launch_bounds__(256)
void edge_pass2(const int* __restrict__ ei, const float* __restrict__ niB,
                const unsigned* __restrict__ maxnjKey,
                float* __restrict__ eb, unsigned* __restrict__ mKey, int En)
{
  const int i = blockIdx.x * blockDim.x + threadIdx.x;
  if (i >= En * 4) return;
  const int e = i >> 2, h = i & 3;
  const int dst = ei[En + e];
  const float maxnj = fkeyinv(maxnjKey[(size_t)dst * 4 + h]) + 1e-8f;
  const float denom = 2.f * (niB[i] + maxnj) + 1e-8f;
  float v = eb[i] / denom;
  v = fminf(fmaxf(v, -10.f), 10.f);
  v = fminf(fmaxf(v, -8.f), 8.f);
  eb[i] = v;
  atomicMax(&mKey[(size_t)dst * 4 + h], fkey(v));
}

// Edge pass 3: exp(e - segmax) and segment sum
__global__ __launch_bounds__(256)
void edge_pass3(const int* __restrict__ ei, const unsigned* __restrict__ mKey,
                float* __restrict__ eb, float* __restrict__ sB, int En)
{
  const int i = blockIdx.x * blockDim.x + threadIdx.x;
  if (i >= En * 4) return;
  const int e = i >> 2, h = i & 3;
  const int dst = ei[En + e];
  const float mx = fkeyinv(mKey[(size_t)dst * 4 + h]);
  const float ex = expf(eb[i] - mx);
  eb[i] = ex;
  __hip_atomic_fetch_add(&sB[(size_t)dst * 4 + h], ex,
                         __ATOMIC_RELAXED, __HIP_MEMORY_SCOPE_AGENT);
}

// Edge pass 4: agg[dst] += alpha * xj  (wave per edge, 4 ch/lane)
__global__ __launch_bounds__(256)
void edge_pass4(const float* __restrict__ xl, const int* __restrict__ ei,
                const float* __restrict__ exB, const float* __restrict__ sB,
                float* __restrict__ agg, int En)
{
  const int tid = threadIdx.x, wave = tid >> 5, lane = tid & 31;
  const int e = blockIdx.x * 8 + wave;
  if (e >= En) return;
  const int src = ei[e], dst = ei[En + e];
  const int h = lane >> 3, c0 = lane * 4;
  const float alpha = exB[(size_t)e * 4 + h] / (sB[(size_t)dst * 4 + h] + 1e-16f);
  const float* xjp = xl + (size_t)src * 128 + c0;
  float* op = agg + (size_t)dst * 128 + c0;
#pragma unroll
  for (int j = 0; j < 4; ++j)
    __hip_atomic_fetch_add(op + j, alpha * xjp[j],
                           __ATOMIC_RELAXED, __HIP_MEMORY_SCOPE_AGENT);
}

// Wave-per-node: +bias, LayerNorm(128), optional +0.1*residual, optional lrelu
__global__ __launch_bounds__(256)
void node_finish(const float* __restrict__ agg, const float* __restrict__ bias,
                 const float* __restrict__ g, const float* __restrict__ b,
                 const float* __restrict__ resid, float* __restrict__ outp,
                 int Nn, int act)
{
  const int tid = threadIdx.x, wave = tid >> 5, lane = tid & 31;
  const int node = blockIdx.x * 8 + wave;
  if (node >= Nn) return;
  const int c0 = lane * 4;
  float v[4];
#pragma unroll
  for (int j = 0; j < 4; ++j)
    v[j] = agg[(size_t)node * 128 + c0 + j] + (bias ? bias[c0 + j] : 0.f);
  float s1 = v[0] + v[1] + v[2] + v[3];
  float s2 = v[0] * v[0] + v[1] * v[1] + v[2] * v[2] + v[3] * v[3];
  for (int msk = 1; msk < 32; msk <<= 1) {
    s1 += __shfl_xor(s1, msk, 32);
    s2 += __shfl_xor(s2, msk, 32);
  }
  const float mean = s1 * (1.f / 128.f);
  const float var  = s2 * (1.f / 128.f) - mean * mean;  // biased, like torch LN
  const float rs   = rsqrtf(var + 1e-5f);
#pragma unroll
  for (int j = 0; j < 4; ++j) {
    float y = (v[j] - mean) * rs * g[c0 + j] + b[c0 + j];
    if (resid) y += 0.1f * resid[(size_t)node * 128 + c0 + j];
    if (act)   y = lrelu(y);
    outp[(size_t)node * 128 + c0 + j] = y;
  }
}

// Final 128 -> 2 projection (thread per node)
__global__ __launch_bounds__(256)
void final_linear(const float* __restrict__ t2, const float* __restrict__ W3,
                  const float* __restrict__ b3, float* __restrict__ out, int Nn)
{
  const int nd = blockIdx.x * blockDim.x + threadIdx.x;
  if (nd >= Nn) return;
  float o0 = b3[0], o1 = b3[1];
#pragma unroll 4
  for (int k = 0; k < 128; ++k) {
    const float v = t2[(size_t)nd * 128 + k];
    o0 += v * W3[k * 2 + 0];
    o1 += v * W3[k * 2 + 1];
  }
  out[nd * 2 + 0] = o0;
  out[nd * 2 + 1] = o1;
}

__global__ void zero_f32(float* p, size_t n) {
  size_t i = (size_t)blockIdx.x * blockDim.x + threadIdx.x;
  const size_t stride = (size_t)gridDim.x * blockDim.x;
  for (; i < n; i += stride) p[i] = 0.f;
}

// ---------------------------------------------------------------------------
extern "C" void kernel_launch(void* const* d_in, const int* in_sizes, int n_in,
                              void* d_out, int out_size, void* d_ws, size_t ws_size,
                              hipStream_t stream) {
  (void)in_sizes; (void)n_in; (void)out_size; (void)ws_size;
  const int N = NNODES, E = NEDGES;

  // Inputs in setup_inputs() insertion order, params flattened recursively.
  const float* x     = (const float*)d_in[0];
  const int*   ei    = (const int*)  d_in[1];
  const float* eattr = (const float*)d_in[2];
  const float* Wl0 = (const float*)d_in[3];  const float* bl0 = (const float*)d_in[4];
  const float* Wr0 = (const float*)d_in[5];  const float* br0 = (const float*)d_in[6];
  const float* We0 = (const float*)d_in[7];  const float* at0 = (const float*)d_in[8];
  const float* bs0 = (const float*)d_in[9];
  const float* Wl1 = (const float*)d_in[10]; const float* bl1 = (const float*)d_in[11];
  const float* Wr1 = (const float*)d_in[12]; const float* br1 = (const float*)d_in[13];
  const float* We1 = (const float*)d_in[14]; const float* at1 = (const float*)d_in[15];
  const float* bs1 = (const float*)d_in[16];
  const float* lg0 = (const float*)d_in[17]; const float* lb0 = (const float*)d_in[18];
  const float* lg1 = (const float*)d_in[19]; const float* lb1 = (const float*)d_in[20];
  const float* plg  = (const float*)d_in[21]; const float* plb  = (const float*)d_in[22];
  const float* W1   = (const float*)d_in[23]; const float* b1   = (const float*)d_in[24];
  const float* bn1g = (const float*)d_in[25]; const float* bn1b = (const float*)d_in[26];
  const float* W2   = (const float*)d_in[27]; const float* b2   = (const float*)d_in[28];
  const float* bn2g = (const float*)d_in[29]; const float* bn2b = (const float*)d_in[30];
  const float* W3   = (const float*)d_in[31]; const float* b3   = (const float*)d_in[32];

  float* ws = (float*)d_ws;
  const size_t NF = (size_t)N * 128;
  const size_t E4 = (size_t)E * 4;
  float* xl   = ws;                      // [N,128]
  float* xr   = xl + NF;                 // [N,128]
  float* hb0  = xr + NF;                 // [N,128]  (later: t1 [N,256])
  float* hb1  = hb0 + NF;                // [N,128]
  float* ebuf = hb1 + NF;                // [E,4]  e_pre -> e -> exp(e)
  float* nib  = ebuf + E4;               // [E,4]
  float* mjK  = nib + E4;                // [N,4] maxnj keys (unsigned)
  float* mK   = mjK + (size_t)N * 4;     // [N,4] softmax-max keys
  float* sS   = mK  + (size_t)N * 4;     // [N,4] softmax sums

  const dim3 B256(256);
  const int gbE8  = E / 8;               // wave per edge
  const int gbE4  = (E * 4) / 256;
  const int gbN8  = N / 8;               // wave per node
  const int gbN16 = N / 16;              // 16-row WMMA tiles

  // ----- layer 0 (K = 64) -----
  zero_f32<<<2048, B256, 0, stream>>>(hb0, NF);
  zero_f32<<<64,   B256, 0, stream>>>(mjK, (size_t)N * 4);
  zero_f32<<<64,   B256, 0, stream>>>(mK,  (size_t)N * 4);
  zero_f32<<<64,   B256, 0, stream>>>(sS,  (size_t)N * 4);
  gemm_wmma<64, 128, false, false><<<dim3(gbN16, 1), B256, 0, stream>>>(
      x, Wl0, bl0, nullptr, nullptr, xl, N);
  gemm_wmma<64, 128, false, false><<<dim3(gbN16, 1), B256, 0, stream>>>(
      x, Wr0, br0, nullptr, nullptr, xr, N);
  edge_pass1<<<gbE8, B256, 0, stream>>>(xl, xr, ei, eattr, We0, at0,
                                        ebuf, nib, (unsigned*)mjK, E);
  edge_pass2<<<gbE4, B256, 0, stream>>>(ei, nib, (const unsigned*)mjK,
                                        ebuf, (unsigned*)mK, E);
  edge_pass3<<<gbE4, B256, 0, stream>>>(ei, (const unsigned*)mK, ebuf, sS, E);
  edge_pass4<<<gbE8, B256, 0, stream>>>(xl, ei, ebuf, sS, hb0, E);
  node_finish<<<gbN8, B256, 0, stream>>>(hb0, bs0, lg0, lb0, nullptr, hb0, N, 1);

  // ----- layer 1 (K = 128, residual = hb0) -----
  zero_f32<<<2048, B256, 0, stream>>>(hb1, NF);
  zero_f32<<<64,   B256, 0, stream>>>(mjK, (size_t)N * 4);
  zero_f32<<<64,   B256, 0, stream>>>(mK,  (size_t)N * 4);
  zero_f32<<<64,   B256, 0, stream>>>(sS,  (size_t)N * 4);
  gemm_wmma<128, 128, false, false><<<dim3(gbN16, 1), B256, 0, stream>>>(
      hb0, Wl1, bl1, nullptr, nullptr, xl, N);
  gemm_wmma<128, 128, false, false><<<dim3(gbN16, 1), B256, 0, stream>>>(
      hb0, Wr1, br1, nullptr, nullptr, xr, N);
  edge_pass1<<<gbE8, B256, 0, stream>>>(xl, xr, ei, eattr, We1, at1,
                                        ebuf, nib, (unsigned*)mjK, E);
  edge_pass2<<<gbE4, B256, 0, stream>>>(ei, nib, (const unsigned*)mjK,
                                        ebuf, (unsigned*)mK, E);
  edge_pass3<<<gbE4, B256, 0, stream>>>(ei, (const unsigned*)mK, ebuf, sS, E);
  edge_pass4<<<gbE8, B256, 0, stream>>>(xl, ei, ebuf, sS, hb1, E);
  node_finish<<<gbN8, B256, 0, stream>>>(hb1, bs1, lg1, lb1, hb0, hb1, N, 1);

  // ----- projection head -----
  // LN(hb1) -> xl ; then hb0..hb1 region is free for t1 [N,256].
  node_finish<<<gbN8, B256, 0, stream>>>(hb1, nullptr, plg, plb, nullptr, xl, N, 0);
  float* t1 = hb0;   // spans hb0..hb1 : N*256 floats
  float* t2 = xr;
  gemm_wmma<128, 256, true, true><<<dim3(gbN16, 2), B256, 0, stream>>>(
      xl, W1, b1, bn1g, bn1b, t1, N);
  gemm_wmma<256, 128, true, true><<<dim3(gbN16, 1), B256, 0, stream>>>(
      t1, W2, b2, bn2g, bn2b, t2, N);
  final_linear<<<(N + 255) / 256, B256, 0, stream>>>(t2, W3, b3, (float*)d_out, N);
}
